// Mamba2_34540126994591
// MI455X (gfx1250) — matrix operand order repaired and checked by reading, not compiled, not run on hardware
//
#include <hip/hip_runtime.h>
#include <hip/hip_bf16.h>
#include <math.h>

// ---------------- problem constants ----------------
#define D_MODEL   2048
#define D_INNER   4096
#define HEADDIM   64
#define D_STATE   64
#define NHEADS    64
#define CONV_DIM  4224          // D_INNER + 2*D_STATE
#define D_IN_PROJ 8384          // 2*D_INNER + 2*D_STATE + NHEADS
#define SEQLEN    1024
#define DT_OFF    8320          // D_INNER + CONV_DIM
#define RMS_EPS   1e-5f

typedef __attribute__((ext_vector_type(16))) _Float16 v16h;
typedef __attribute__((ext_vector_type(8)))  _Float16 v8h;
typedef __attribute__((ext_vector_type(8)))  float    v8f;

// ---------------- fp32 -> fp16 convert ----------------
__global__ void cvt_f32_f16(const float* __restrict__ src,
                            _Float16* __restrict__ dst, size_t n) {
  size_t i = (size_t)blockIdx.x * blockDim.x + threadIdx.x;
  size_t stride = (size_t)gridDim.x * blockDim.x;
  for (; i < n; i += stride) dst[i] = (_Float16)src[i];
}

// ---------------- WMMA GEMM:  C[M,N] = A[M,K] * B[N,K]^T ----------------
// A, B row-major with K contiguous (both operands K-major -> matches CDNA5
// WMMA per-lane layouts; plain b128 loads, no transpose needed).
// One wave -> 16 rows x 64 cols (4 accumulators).  4 waves/block -> 64x64 tile.
__global__ __launch_bounds__(128)
void wmma_gemm_nt(const _Float16* __restrict__ A, const _Float16* __restrict__ B,
                  float* __restrict__ C, int M, int N, int K) {
  const int lane = threadIdx.x & 31;
  const int wave = threadIdx.x >> 5;
  const int m0 = blockIdx.y * 64 + wave * 16;
  const int n0 = blockIdx.x * 64;
  if (m0 >= M || n0 >= N) return;

  // A fragment: lanes 0-15 hold M=lane, K = k0+0..15
  //             lanes 16-31 hold M=lane-16, K = k0+16..31
  const int arow  = lane & 15;
  const int ahalf = (lane >> 4) * 16;
  // B fragment: lanes 0-15 col=lane K=k0+0..15; lanes 16-31 col=lane-16 K=k0+16..31
  const int bcol  = lane & 15;
  const int bhalf = (lane >> 4) * 16;

  const _Float16* aptr = A + (size_t)(m0 + arow) * K + ahalf;
  const _Float16* b0 = B + (size_t)(n0 +  0 + bcol) * K + bhalf;
  const _Float16* b1 = B + (size_t)(n0 + 16 + bcol) * K + bhalf;
  const _Float16* b2 = B + (size_t)(n0 + 32 + bcol) * K + bhalf;
  const _Float16* b3 = B + (size_t)(n0 + 48 + bcol) * K + bhalf;

  v8f c0 = {}, c1 = {}, c2 = {}, c3 = {};

  for (int k0 = 0; k0 < K; k0 += 32) {
    // prefetch next K-slab (speculative; dropped if OOB)
    __builtin_prefetch(aptr + k0 + 512, 0, 0);
    __builtin_prefetch(b0 + k0 + 512, 0, 0);
    __builtin_prefetch(b2 + k0 + 512, 0, 0);

    v8h alo = *(const v8h*)(aptr + k0);
    v8h ahi = *(const v8h*)(aptr + k0 + 8);
    v16h a = __builtin_shufflevector(alo, ahi,
             0,1,2,3,4,5,6,7,8,9,10,11,12,13,14,15);

    v8h t0, t1;
    t0 = *(const v8h*)(b0 + k0); t1 = *(const v8h*)(b0 + k0 + 8);
    v16h bb0 = __builtin_shufflevector(t0, t1,
              0,1,2,3,4,5,6,7,8,9,10,11,12,13,14,15);
    t0 = *(const v8h*)(b1 + k0); t1 = *(const v8h*)(b1 + k0 + 8);
    v16h bb1 = __builtin_shufflevector(t0, t1,
              0,1,2,3,4,5,6,7,8,9,10,11,12,13,14,15);
    t0 = *(const v8h*)(b2 + k0); t1 = *(const v8h*)(b2 + k0 + 8);
    v16h bb2 = __builtin_shufflevector(t0, t1,
              0,1,2,3,4,5,6,7,8,9,10,11,12,13,14,15);
    t0 = *(const v8h*)(b3 + k0); t1 = *(const v8h*)(b3 + k0 + 8);
    v16h bb3 = __builtin_shufflevector(t0, t1,
              0,1,2,3,4,5,6,7,8,9,10,11,12,13,14,15);

    c0 = __builtin_amdgcn_wmma_f32_16x16x32_f16(false, a, false, bb0, (short)0, c0, false, false);
    c1 = __builtin_amdgcn_wmma_f32_16x16x32_f16(false, a, false, bb1, (short)0, c1, false, false);
    c2 = __builtin_amdgcn_wmma_f32_16x16x32_f16(false, a, false, bb2, (short)0, c2, false, false);
    c3 = __builtin_amdgcn_wmma_f32_16x16x32_f16(false, a, false, bb3, (short)0, c3, false, false);
  }

  // D layout: lanes 0-15: N=lane, VGPR r -> M=r ; lanes 16-31: N=lane-16, M=8+r
  const int mlo = (lane >> 4) * 8;
#pragma unroll
  for (int r = 0; r < 8; ++r) {
    float* crow = C + (size_t)(m0 + mlo + r) * N + n0 + bcol;
    crow[0]  = c0[r];
    crow[16] = c1[r];
    crow[32] = c2[r];
    crow[48] = c3[r];
  }
}

// ---------------- depthwise causal conv (k=4) + SiLU ----------------
__global__ void conv_silu_kernel(const float* __restrict__ zx,
                                 const float* __restrict__ conv_w,
                                 const float* __restrict__ conv_b,
                                 float* __restrict__ xBCc) {
  int idx = blockIdx.x * blockDim.x + threadIdx.x;
  if (idx >= SEQLEN * CONV_DIM) return;
  int c = idx % CONV_DIM;
  int t = idx / CONV_DIM;
  const float* col = zx + D_INNER + c;
  float acc = conv_b[c];
#pragma unroll
  for (int k = 0; k < 4; ++k) {
    int tt = t - 3 + k;
    if (tt >= 0) acc += col[(size_t)tt * D_IN_PROJ] * conv_w[c * 4 + k];
  }
  xBCc[idx] = acc / (1.0f + expf(-acc));   // silu
}

// ---------------- dt = softplus(raw + bias);  dA = exp(dt * -exp(A_log)) ----
__global__ void dt_kernel(const float* __restrict__ zx,
                          const float* __restrict__ dt_bias,
                          const float* __restrict__ A_log,
                          float* __restrict__ dt_sp, float* __restrict__ dAo) {
  int idx = blockIdx.x * blockDim.x + threadIdx.x;
  if (idx >= SEQLEN * NHEADS) return;
  int h = idx & 63;
  int t = idx >> 6;
  float raw = zx[(size_t)t * D_IN_PROJ + DT_OFF + h] + dt_bias[h];
  float dt = raw > 20.f ? raw : log1pf(expf(raw));
  float Av = -expf(A_log[h]);
  dt_sp[idx] = dt;
  dAo[idx]   = expf(dt * Av);
}

// ---------------- sequential SSM scan, one workgroup per head -------------
// thread owns p = tid>>2, n-chunk = (tid&3)*16  ->  16 state elems in regs
__global__ __launch_bounds__(256)
void scan_kernel(const float* __restrict__ xBCc,
                 const float* __restrict__ dt_sp,
                 const float* __restrict__ dAo,
                 const float* __restrict__ Dp,
                 float* __restrict__ y) {
  const int head = blockIdx.x;
  const int tid  = threadIdx.x;
  const int p    = tid >> 2;
  const int nc   = (tid & 3) << 4;

  float h[16];
#pragma unroll
  for (int i = 0; i < 16; ++i) h[i] = 0.f;
  const float Dv = Dp[head];

  for (int t = 0; t < SEQLEN; ++t) {
    const float* row = xBCc + (size_t)t * CONV_DIM;
    const float dAv = dAo[t * NHEADS + head];
    const float dtv = dt_sp[t * NHEADS + head];
    const float xv  = row[head * HEADDIM + p];
    const float dx  = dtv * xv;

    const float4* B4 = (const float4*)(row + D_INNER + nc);
    const float4* C4 = (const float4*)(row + D_INNER + D_STATE + nc);
    float yp = 0.f;
#pragma unroll
    for (int j = 0; j < 4; ++j) {
      float4 b = B4[j];
      float4 c = C4[j];
      h[j*4+0] = h[j*4+0] * dAv + dx * b.x;  yp += h[j*4+0] * c.x;
      h[j*4+1] = h[j*4+1] * dAv + dx * b.y;  yp += h[j*4+1] * c.y;
      h[j*4+2] = h[j*4+2] * dAv + dx * b.z;  yp += h[j*4+2] * c.z;
      h[j*4+3] = h[j*4+3] * dAv + dx * b.w;  yp += h[j*4+3] * c.w;
    }
    // reduce across the 4 consecutive lanes that share p (wave32-safe)
    yp += __shfl_xor(yp, 1, 32);
    yp += __shfl_xor(yp, 2, 32);
    if ((tid & 3) == 0)
      y[(size_t)t * D_INNER + head * HEADDIM + p] = yp + Dv * xv;
  }
}

// ---------------- y * silu(z), RMS norm, emit f16 for out_proj -------------
__global__ __launch_bounds__(256)
void gate_norm_kernel(const float* __restrict__ y,
                      const float* __restrict__ zx,
                      const float* __restrict__ norm_w,
                      _Float16* __restrict__ yh) {
  const int t   = blockIdx.x;
  const int tid = threadIdx.x;
  const float* yt = y  + (size_t)t * D_INNER;
  const float* zt = zx + (size_t)t * D_IN_PROJ;

  float vals[16];
  float ss = 0.f;
#pragma unroll
  for (int i = 0; i < 16; ++i) {
    int c = i * 256 + tid;
    float z = zt[c];
    float g = z / (1.0f + expf(-z));   // silu(z)
    float v = yt[c] * g;
    vals[i] = v;
    ss += v * v;
  }

  __shared__ float red[256];
  red[tid] = ss;
  __syncthreads();
  for (int s = 128; s > 0; s >>= 1) {
    if (tid < s) red[tid] += red[tid + s];
    __syncthreads();
  }
  const float rinv = rsqrtf(red[0] * (1.0f / D_INNER) + RMS_EPS);

#pragma unroll
  for (int i = 0; i < 16; ++i) {
    int c = i * 256 + tid;
    yh[(size_t)t * D_INNER + c] = (_Float16)(vals[i] * rinv * norm_w[c]);
  }
}

// ---------------- host launch ----------------
extern "C" void kernel_launch(void* const* d_in, const int* in_sizes, int n_in,
                              void* d_out, int out_size, void* d_ws, size_t ws_size,
                              hipStream_t stream) {
  const float* x       = (const float*)d_in[0];
  const float* W_in    = (const float*)d_in[1];
  const float* conv_w  = (const float*)d_in[2];
  const float* conv_b  = (const float*)d_in[3];
  const float* dt_bias = (const float*)d_in[4];
  const float* A_log   = (const float*)d_in[5];
  const float* Dp      = (const float*)d_in[6];
  const float* norm_w  = (const float*)d_in[7];
  const float* W_out   = (const float*)d_in[8];
  float* out = (float*)d_out;

  // workspace carve-up (256B aligned)
  char* ws = (char*)d_ws;
  size_t off = 0;
  auto carve = [&](size_t bytes) -> void* {
    void* p = ws + off;
    off += (bytes + 255) & ~(size_t)255;
    return p;
  };
  _Float16* W_in_h  = (_Float16*)carve((size_t)D_IN_PROJ * D_MODEL * 2);
  _Float16* W_out_h = (_Float16*)carve((size_t)D_MODEL * D_INNER * 2);
  _Float16* x_h     = (_Float16*)carve((size_t)SEQLEN * D_MODEL * 2);
  float*    zx      = (float*)   carve((size_t)SEQLEN * D_IN_PROJ * 4);
  float*    xBCc    = (float*)   carve((size_t)SEQLEN * CONV_DIM * 4);
  float*    dt_sp   = (float*)   carve((size_t)SEQLEN * NHEADS * 4);
  float*    dAo     = (float*)   carve((size_t)SEQLEN * NHEADS * 4);
  float*    yb      = (float*)   carve((size_t)SEQLEN * D_INNER * 4);
  _Float16* y_h     = (_Float16*)carve((size_t)SEQLEN * D_INNER * 2);
  (void)ws_size; (void)n_in; (void)in_sizes; (void)out_size;

  // 1) convert weights/activations to f16
  cvt_f32_f16<<<2048, 256, 0, stream>>>(W_in,  W_in_h,  (size_t)D_IN_PROJ * D_MODEL);
  cvt_f32_f16<<<2048, 256, 0, stream>>>(W_out, W_out_h, (size_t)D_MODEL * D_INNER);
  cvt_f32_f16<<<1024, 256, 0, stream>>>(x,     x_h,     (size_t)SEQLEN * D_MODEL);

  // 2) in_proj GEMM: zx[1024,8384] = x[1024,2048] @ W_in[8384,2048]^T
  {
    dim3 grid(D_IN_PROJ / 64, SEQLEN / 64);
    wmma_gemm_nt<<<grid, 128, 0, stream>>>(x_h, W_in_h, zx, SEQLEN, D_IN_PROJ, D_MODEL);
  }

  // 3) depthwise conv + silu
  conv_silu_kernel<<<(SEQLEN * CONV_DIM + 255) / 256, 256, 0, stream>>>(
      zx, conv_w, conv_b, xBCc);

  // 4) dt softplus + dA
  dt_kernel<<<(SEQLEN * NHEADS + 255) / 256, 256, 0, stream>>>(
      zx, dt_bias, A_log, dt_sp, dAo);

  // 5) SSM scan (one workgroup per head)
  scan_kernel<<<NHEADS, 256, 0, stream>>>(xBCc, dt_sp, dAo, Dp, yb);

  // 6) gating + RMS norm -> f16 activations
  gate_norm_kernel<<<SEQLEN, 256, 0, stream>>>(yb, zx, norm_w, y_h);

  // 7) out_proj GEMM: out[1024,2048] = y[1024,4096] @ W_out[2048,4096]^T
  {
    dim3 grid(D_MODEL / 64, SEQLEN / 64);
    wmma_gemm_nt<<<grid, 128, 0, stream>>>(y_h, W_out_h, out, SEQLEN, D_MODEL, D_INNER);
  }
}
